// spike_dense_test_denri_wotanh_R_xor_76819785056529
// MI455X (gfx1250) — matrix operand order repaired and checked by reading, not compiled, not run on hardware
//
#include <hip/hip_runtime.h>

// Problem constants (match the reference).
#define B_SZ   64
#define IN_SZ  8192
#define OUT_SZ 2048
#define BR_SZ  4
#define CHUNK  (IN_SZ / BR_SZ)   // 2048 = K per branch

typedef __attribute__((ext_vector_type(16))) __bf16 bf16x16;
typedef __attribute__((ext_vector_type(8)))  float  f32x8;
typedef __attribute__((ext_vector_type(4)))  float  f32x4;

__device__ __forceinline__ float sigmoidf_fast(float x) {
    return 1.0f / (1.0f + __expf(-x));
}

// Pack 16 floats (K order) into a bf16x16 WMMA fragment.
__device__ __forceinline__ bf16x16 cvt_frag(f32x4 x0, f32x4 x1, f32x4 x2, f32x4 x3) {
    bf16x16 r;
#pragma unroll
    for (int i = 0; i < 4; ++i) {
        r[i]      = (__bf16)x0[i];
        r[4 + i]  = (__bf16)x1[i];
        r[8 + i]  = (__bf16)x2[i];
        r[12 + i] = (__bf16)x3[i];
    }
    return r;
}

// A fragment (contiguous K order, matching B), a_ptr already includes +kgrp*16:
//   elements 0..15 <- K = kgrp*16 + 0..15
__device__ __forceinline__ bf16x16 load_a_frag(const float* __restrict__ a_ptr, int kb) {
    const f32x4* p0 = (const f32x4*)(a_ptr + kb);
    const f32x4* p1 = (const f32x4*)(a_ptr + kb + 8);
    return cvt_frag(p0[0], p0[1], p1[0], p1[1]);
}

// B fragment (contiguous K order, matching A), b_ptr already includes +kgrp*16:
//   elements 0..15 <- K = kgrp*16 + 0..15 (16 contiguous floats per lane)
__device__ __forceinline__ bf16x16 load_b_frag(const float* __restrict__ b_ptr, int kb) {
    const f32x4* p = (const f32x4*)(b_ptr + kb);
    return cvt_frag(p[0], p[1], p[2], p[3]);
}

__device__ __forceinline__ f32x8 wmma_bf16(bf16x16 a, bf16x16 b, f32x8 acc) {
    return __builtin_amdgcn_wmma_f32_16x16x32_bf16(
        /*neg_a=*/false, a, /*neg_b=*/false, b,
        /*c_mod=*/(short)0, acc, /*reuse_a=*/false, /*reuse_b=*/false);
}

__global__ __launch_bounds__(128)
void snn_denri_fused_kernel(const float* __restrict__ input_spike, // [B, IN]
                            const float* __restrict__ mem,         // [B, OUT]
                            const float* __restrict__ spike,       // [B, OUT]
                            const float* __restrict__ d_input,     // [B, OUT, BR]
                            const float* __restrict__ v_th,        // [B, OUT]
                            const float* __restrict__ W,           // [OUT*BR, IN]
                            const float* __restrict__ bias,        // [OUT*BR]
                            const float* __restrict__ tau_m,       // [OUT]
                            const float* __restrict__ tau_n,       // [OUT, BR]
                            float* __restrict__ out_mem,           // [B, OUT]
                            float* __restrict__ out_spike)         // [B, OUT]
{
    const int lane = threadIdx.x & 31;
    const int br   = threadIdx.x >> 5;     // one wave per dendritic branch
    const int n    = lane & 15;            // column within a 16-wide N subtile
    const int kgrp = lane >> 4;            // 0/1: K half-select per ISA layout

    const int col_base = blockIdx.x * 32;  // 2 x 16 output-neuron subtiles
    const int row_base = blockIdx.y * 16;  // batch tile

    // A: one fragment shared by both N subtiles (halves L2 A-traffic + cvt work).
    const float* a_ptr = input_spike
                       + (size_t)(row_base + n) * IN_SZ + (size_t)br * CHUNK + kgrp * 16;
    // B: W row ((o)*BR + br), diagonal-block column slice of branch br.
    const float* b_ptr0 = W + (size_t)((col_base +      n) * BR_SZ + br) * IN_SZ
                            + (size_t)br * CHUNK + kgrp * 16;
    const float* b_ptr1 = W + (size_t)((col_base + 16 + n) * BR_SZ + br) * IN_SZ
                            + (size_t)br * CHUNK + kgrp * 16;

    f32x8 acc0 = {0.f, 0.f, 0.f, 0.f, 0.f, 0.f, 0.f, 0.f};
    f32x8 acc1 = {0.f, 0.f, 0.f, 0.f, 0.f, 0.f, 0.f, 0.f};

    // K loop: one shared A fragment, two B fragments, two WMMAs into
    // independent accumulators (no WMMA->WMMA RAW on the same D).
    for (int kb = 0; kb < CHUNK; kb += 32) {
        // Prefetch the streaming W rows ~1KB ahead (speculative, safe at end).
        __builtin_prefetch(b_ptr0 + kb + 256, 0, 1);
        __builtin_prefetch(b_ptr1 + kb + 256, 0, 1);
        bf16x16 af = load_a_frag(a_ptr, kb);
        acc0 = wmma_bf16(af, load_b_frag(b_ptr0, kb), acc0);
        acc1 = wmma_bf16(af, load_b_frag(b_ptr1, kb), acc1);
    }

    // ---- per-branch epilogue: bias + dendritic decay mix, both subtiles ----
    __shared__ float lds_d[BR_SZ][2][32][8];   // 8 KB branch-partial dendrites

#pragma unroll
    for (int t = 0; t < 2; ++t) {
        const int   o      = col_base + t * 16 + n;
        const float bias_v = bias[(size_t)o * BR_SZ + br];
        const float beta   = sigmoidf_fast(tau_n[(size_t)o * BR_SZ + br]);
        const f32x8 acc    = t ? acc1 : acc0;
#pragma unroll
        for (int v = 0; v < 8; ++v) {
            const int   row = row_base + v + 8 * kgrp;  // C/D: m = v + 8*(lane/16)
            const float y   = acc[v] + bias_v;
            const float din = d_input[((size_t)row * OUT_SZ + o) * BR_SZ + br];
            lds_d[br][t][lane][v] = beta * din + (1.0f - beta) * y;
        }
    }
    __syncthreads();

    // ---- branch reduction + LIF + spike: wave t finalizes subtile t ----
    if (br < 2) {
        const int   t = br;
        const int   o = col_base + t * 16 + n;
        const float alpha = sigmoidf_fast(tau_m[o]);
#pragma unroll
        for (int v = 0; v < 8; ++v) {
            const int    row = row_base + v + 8 * kgrp;
            const size_t idx = (size_t)row * OUT_SZ + o;
            const float  l   = lds_d[0][t][lane][v] + lds_d[1][t][lane][v]
                             + lds_d[2][t][lane][v] + lds_d[3][t][lane][v];
            const float  vt  = v_th[idx];
            const float  m_new = mem[idx] * alpha + (1.0f - alpha) * l - vt * spike[idx];
            out_mem[idx]   = m_new;
            out_spike[idx] = (m_new - vt) > 0.0f ? 1.0f : 0.0f;
        }
    }
}

extern "C" void kernel_launch(void* const* d_in, const int* in_sizes, int n_in,
                              void* d_out, int out_size, void* d_ws, size_t ws_size,
                              hipStream_t stream) {
    (void)in_sizes; (void)n_in; (void)out_size; (void)d_ws; (void)ws_size;
    const float* input_spike = (const float*)d_in[0];
    const float* mem         = (const float*)d_in[1];
    const float* spike       = (const float*)d_in[2];
    const float* d_input     = (const float*)d_in[3];
    const float* v_th        = (const float*)d_in[4];
    const float* W           = (const float*)d_in[5];
    const float* bias        = (const float*)d_in[6];
    const float* tau_m       = (const float*)d_in[7];
    const float* tau_n       = (const float*)d_in[8];

    float* out_mem   = (float*)d_out;
    float* out_spike = out_mem + (size_t)B_SZ * OUT_SZ;

    dim3 grid(OUT_SZ / 32, B_SZ / 16);  // 64 N-blocks x 4 M-tiles
    dim3 block(128);                    // 4 wave32s = 4 branches
    snn_denri_fused_kernel<<<grid, block, 0, stream>>>(
        input_spike, mem, spike, d_input, v_th, W, bias, tau_m, tau_n,
        out_mem, out_spike);
}
